// LigerFusedLinearORPOLoss_35150012350893
// MI455X (gfx1250) — matrix operand order repaired and checked by reading, not compiled, not run on hardware
//
#include <hip/hip_runtime.h>
#include <cstdint>
#include <cstddef>

// ---------------------------------------------------------------------------
// Fused Linear + ORPO loss for MI455X (gfx1250, wave32, WMMA).
// B=8, T=1024, H=2048, V=32000.
// bf16 WMMA GEMM fused with online logsumexp; fp32 target-logit dot; scalar
// finalize. This revision: explicit ping-pong double-buffered k-loop with
// unrolling disabled so the compiler can neither fully unroll nor hoist the
// LDS A-fragment loads (round 2 showed full unroll -> LICM -> 4KB/lane
// scratch spills). Each WMMA now waits only on its own buffer's loads while
// the other buffer's loads are in flight.
// ---------------------------------------------------------------------------

#define IGNORE_INDEX (-100)
#define BETA 0.1f

#define BB 8
#define TT 1024
#define HH 2048
#define VV 32000
#define ROWS (BB * TT)            // 8192
#define MT 32                     // rows per block (A tile)
#define ROW_TILES (ROWS / MT)     // 256
#define CHUNK 1024                // logit columns per block
#define NCHUNK 32                 // 32*1024 = 32768 >= V (guarded)
#define NSLOT (NCHUNK * 8)        // partial-LSE slots per row (8 N-waves/chunk)
#define PADH (HH + 8)             // padded LDS row stride (ushorts): 4112 B
                                  // 4112 % 256 == 16 -> 4-bank rotation/row,
                                  // 16-lane b128 A-load covers all 64 banks

typedef __attribute__((ext_vector_type(16))) __bf16 v16bf;
typedef __attribute__((ext_vector_type(8)))  float  v8f;

union Frag16 { v16bf v; unsigned short u[16]; uint4 q[2]; };
struct Bufs { Frag16 a0, a1, b; };

__device__ __forceinline__ unsigned short f32_to_bf16_rne(float f) {
  unsigned int u = __float_as_uint(f);
  u += 0x7FFFu + ((u >> 16) & 1u);      // round to nearest even
  return (unsigned short)(u >> 16);
}

__device__ __forceinline__ void load_a(Frag16& f, const unsigned short* Atile,
                                       int mrow, int ka) {
  const unsigned short* ap = Atile + mrow * PADH + ka;
  f.q[0] = *(const uint4*)(ap);
  f.q[1] = *(const uint4*)(ap + 16);
}
__device__ __forceinline__ void load_b(Frag16& f, const unsigned short* Wrow,
                                       int kb) {
  f.q[0] = *(const uint4*)(Wrow + kb);
  f.q[1] = *(const uint4*)(Wrow + kb + 16);
}
__device__ __forceinline__ void load_bufs(Bufs& f, const unsigned short* Atile,
                                          const unsigned short* Wrow,
                                          int lane16, int halfSel, int k) {
  load_a(f.a0, Atile, lane16,      k + halfSel * 8);
  load_a(f.a1, Atile, 16 + lane16, k + halfSel * 8);
  load_b(f.b, Wrow, k + halfSel * 8);
}
__device__ __forceinline__ void mma_pair(v8f& acc0, v8f& acc1, const Bufs& f) {
  acc0 = __builtin_amdgcn_wmma_f32_16x16x32_bf16(
      false, f.a0.v, false, f.b.v, (short)0, acc0, false, false);
  acc1 = __builtin_amdgcn_wmma_f32_16x16x32_bf16(
      false, f.a1.v, false, f.b.v, (short)0, acc1, false, false);
}
__device__ __forceinline__ void lse_update(float& m, float& s, float val) {
  if (val > m) { s = s * __expf(m - val) + 1.f; m = val; }
  else         { s += __expf(val - m); }
}

// ---------------------------------------------------------------- K0: convert
__global__ void cvt_f32_bf16_kernel(const float* __restrict__ src,
                                    unsigned short* __restrict__ dst, int n) {
  int i = blockIdx.x * blockDim.x + threadIdx.x;
  int stride = gridDim.x * blockDim.x;
  for (; i < n; i += stride) dst[i] = f32_to_bf16_rne(src[i]);
}

// ------------------------------------------------- K1: exact fp32 target logit
__global__ void target_logit_kernel(const float* __restrict__ W,
                                    const float* __restrict__ X,
                                    const float* __restrict__ bias,
                                    const int* __restrict__ target,
                                    float* __restrict__ tl) {
  __shared__ float red[256];
  const int row = blockIdx.x;
  int t = target[row];
  int tc = t < 0 ? 0 : t;                 // jnp.clip(target, 0)
  const float* xr = X + (size_t)row * HH;
  const float* wr = W + (size_t)tc * HH;
  float acc = 0.f;
  for (int h = threadIdx.x; h < HH; h += 256) acc += xr[h] * wr[h];
  red[threadIdx.x] = acc;
  __syncthreads();
  for (int off = 128; off > 0; off >>= 1) {
    if (threadIdx.x < off) red[threadIdx.x] += red[threadIdx.x + off];
    __syncthreads();
  }
  if (threadIdx.x == 0) tl[row] = red[0] + bias[tc];
}

// ---------------------------------- K2: WMMA GEMM + online logsumexp partials
__global__ void __launch_bounds__(256)
orpo_lse_gemm_kernel(const unsigned short* __restrict__ W,   // [V][H] bf16
                     const unsigned short* __restrict__ X,   // [ROWS][H] bf16
                     const float* __restrict__ bias,         // [V]
                     float* __restrict__ pmax,               // [ROWS][NSLOT]
                     float* __restrict__ psum)               // [ROWS][NSLOT]
{
  extern __shared__ unsigned short Atile[];   // MT x PADH bf16 = 128.5 KB LDS
  const int tid   = threadIdx.x;
  const int lane  = tid & 31;
  const int wave  = tid >> 5;       // 0..7 : pure-N split, 16 cols per wave
  const int rowBase   = blockIdx.x * MT;
  const int chunkBase = blockIdx.y * CHUNK;

  // cooperative load of A tile (MT x HH bf16) into padded LDS, b128 copies
  {
    const uint4* src = (const uint4*)(X + (size_t)rowBase * HH);
    uint4* dst = (uint4*)Atile;
    for (int i = tid; i < MT * (HH / 8); i += 256) {
      const int r = i >> 8;               // HH/8 = 256 uint4 per row
      const int c = i & 255;
      dst[r * (PADH / 8) + c] = src[i];   // PADH/8 = 257
    }
  }
  __syncthreads();

  const int halfSel = lane >> 4;        // 0: lanes 0-15, 1: lanes 16-31
  const int lane16  = lane & 15;

  // per-lane online softmax stats: rows 0-15 via acc0, rows 16-31 via acc1
  float m[16], s[16];
  #pragma unroll
  for (int i = 0; i < 16; ++i) { m[i] = -__builtin_inff(); s[i] = 0.f; }

  #pragma clang loop unroll(disable)
  for (int niter = 0; niter < CHUNK / 128; ++niter) {
    const int col = chunkBase + niter * 128 + wave * 16 + lane16;
    const int vcl = col < VV ? col : (VV - 1);    // clamp for safe loads
    const unsigned short* Wrow = W + (size_t)vcl * HH;

    v8f acc0 = {}, acc1 = {};
    Bufs f0, f1;
    load_bufs(f0, Atile, Wrow, lane16, halfSel, 0);
    // ping-pong double buffer, two k-steps (2x32) per iteration, no rotation
    #pragma clang loop unroll(disable)
    for (int k = 0; k < HH - 64; k += 64) {
      load_bufs(f1, Atile, Wrow, lane16, halfSel, k + 32);
      __builtin_prefetch(Wrow + k + 256, 0, 1);   // global_prefetch_b8
      mma_pair(acc0, acc1, f0);                   // waits only on f0's loads
      load_bufs(f0, Atile, Wrow, lane16, halfSel, k + 64);
      mma_pair(acc0, acc1, f1);                   // waits only on f1's loads
    }
    load_bufs(f1, Atile, Wrow, lane16, halfSel, HH - 32);
    mma_pair(acc0, acc1, f0);
    mma_pair(acc0, acc1, f1);

    // C layout: VGPR i, lanes 0-15 -> M=i, lanes 16-31 -> M=i+8; N = lane16
    const float bv = bias[vcl];
    if (col < VV) {
      #pragma unroll
      for (int i = 0; i < 8; ++i) lse_update(m[i],     s[i],     acc0[i] + bv);
      #pragma unroll
      for (int i = 0; i < 8; ++i) lse_update(m[8 + i], s[8 + i], acc1[i] + bv);
    }
  }

  // merge the 16 column-lanes of each half-wave (logsumexp combine)
  #pragma unroll
  for (int i = 0; i < 16; ++i) {
    #pragma unroll
    for (int msk = 1; msk < 16; msk <<= 1) {
      float om = __shfl_xor(m[i], msk, 32);
      float os = __shfl_xor(s[i], msk, 32);
      float nm = fmaxf(m[i], om);
      float ns = 0.f;
      if (m[i] != -__builtin_inff()) ns += s[i] * __expf(m[i] - nm);
      if (om   != -__builtin_inff()) ns += os   * __expf(om   - nm);
      m[i] = nm; s[i] = ns;
    }
  }
  if (lane16 == 0) {   // lane 0: rows {0-7,16-23}; lane 16: rows {8-15,24-31}
    const int slot = blockIdx.y * 8 + wave;
    #pragma unroll
    for (int i = 0; i < 16; ++i) {
      const int tileRow = (i < 8 ? 0 : 16) + halfSel * 8 + (i & 7);
      const int row = rowBase + tileRow;
      pmax[(size_t)row * NSLOT + slot] = m[i];
      psum[(size_t)row * NSLOT + slot] = s[i];
    }
  }
}

// ------------------------------------- K3: per-row LSE merge -> per-token logp
__global__ void lse_reduce_kernel(const float* __restrict__ pmax,
                                  const float* __restrict__ psum,
                                  const float* __restrict__ tl,
                                  const int* __restrict__ target,
                                  float* __restrict__ ptl) {
  const int row = blockIdx.x * blockDim.x + threadIdx.x;
  if (row >= ROWS) return;
  float m = -__builtin_inff(), s = 0.f;
  const float* pm = pmax + (size_t)row * NSLOT;
  const float* ps = psum + (size_t)row * NSLOT;
  for (int i = 0; i < NSLOT; ++i) {
    float om = pm[i], os = ps[i];
    float nm = fmaxf(m, om);
    float ns = 0.f;
    if (m  != -__builtin_inff()) ns += s  * __expf(m  - nm);
    if (om != -__builtin_inff()) ns += os * __expf(om - nm);
    m = nm; s = ns;
  }
  const float lse = m + logf(s);
  const int t = target[row];
  ptl[row] = (t != IGNORE_INDEX) ? (tl[row] - lse) : 0.f;
}

// -------------------------------------------------- K4: ORPO scalar finalize
__global__ void finalize_kernel(const float* __restrict__ ptl,
                                const int* __restrict__ target,
                                float* __restrict__ out) {
  __shared__ float red[1024];
  __shared__ float ssum[BB], scnt[BB];
  const int tid = threadIdx.x;
  for (int b = 0; b < BB; ++b) {
    const int   t   = target[b * TT + tid];
    const float msk = (t != IGNORE_INDEX) ? 1.f : 0.f;
    red[tid] = ptl[b * TT + tid];       // already 0 where masked
    __syncthreads();
    for (int off = 512; off > 0; off >>= 1) {
      if (tid < off) red[tid] += red[tid + off];
      __syncthreads();
    }
    if (tid == 0) ssum[b] = red[0];
    __syncthreads();
    red[tid] = msk;
    __syncthreads();
    for (int off = 512; off > 0; off >>= 1) {
      if (tid < off) red[tid] += red[tid + off];
      __syncthreads();
    }
    if (tid == 0) scnt[b] = red[0];
    __syncthreads();
  }
  if (tid == 0) {
    float seq[BB];
    for (int b = 0; b < BB; ++b) seq[b] = ssum[b] / fmaxf(scnt[b], 1.f);
    float csum = 0.f, ccnt = 0.f;
    for (int b = 0; b < BB / 2; ++b) { csum += ssum[b]; ccnt += scnt[b]; }
    const float nll = -csum / fmaxf(ccnt, 1.f);
    float orv = 0.f;
    for (int p = 0; p < BB / 2; ++p) {
      const float c = seq[p], r = seq[p + BB / 2];
      const float lo = (c - r) - (log1pf(-expf(c)) - log1pf(-expf(r)));
      const float lsig = (lo < 0.f) ? (lo - log1pf(expf(lo)))
                                    : (-log1pf(expf(-lo)));
      orv += lsig;
    }
    out[0] = nll - BETA * orv / (float)(BB / 2);
  }
}

// ---------------------------------------------------------------------------
extern "C" void kernel_launch(void* const* d_in, const int* in_sizes, int n_in,
                              void* d_out, int out_size, void* d_ws, size_t ws_size,
                              hipStream_t stream) {
  (void)in_sizes; (void)n_in; (void)out_size; (void)ws_size;
  const float* W32  = (const float*)d_in[0];   // lin_weight [V, H]
  const float* X32  = (const float*)d_in[1];   // _input    [B, T, H]
  const int*   tgt  = (const int*)d_in[2];     // target    [B, T]
  const float* bias = (const float*)d_in[3];   // bias      [V]
  float* out = (float*)d_out;

  // workspace layout (~182 MB)
  char* ws = (char*)d_ws;
  size_t offW   = 0;                                       // bf16 weight
  size_t offX   = offW   + (size_t)VV * HH * 2;            // bf16 input
  size_t offTL  = offX   + (size_t)ROWS * HH * 2;          // target logits
  size_t offPM  = offTL  + (size_t)ROWS * 4;               // partial max
  size_t offPS  = offPM  + (size_t)ROWS * NSLOT * 4;       // partial sumexp
  size_t offPTL = offPS  + (size_t)ROWS * NSLOT * 4;       // per-token logp
  unsigned short* Wb = (unsigned short*)(ws + offW);
  unsigned short* Xb = (unsigned short*)(ws + offX);
  float* tl   = (float*)(ws + offTL);
  float* pmax = (float*)(ws + offPM);
  float* psum = (float*)(ws + offPS);
  float* ptl  = (float*)(ws + offPTL);

  cvt_f32_bf16_kernel<<<4096, 256, 0, stream>>>(W32, Wb, VV * HH);
  cvt_f32_bf16_kernel<<<2048, 256, 0, stream>>>(X32, Xb, ROWS * HH);
  target_logit_kernel<<<ROWS, 256, 0, stream>>>(W32, X32, bias, tgt, tl);
  dim3 grid(ROW_TILES, NCHUNK);
  orpo_lse_gemm_kernel<<<grid, 256, MT * PADH * 2, stream>>>(Wb, Xb, bias,
                                                             pmax, psum);
  lse_reduce_kernel<<<ROWS / 256, 256, 0, stream>>>(pmax, psum, tl, tgt, ptl);
  finalize_kernel<<<1, 1024, 0, stream>>>(ptl, tgt, out);
}
